// TripletLoss_24034636989223
// MI455X (gfx1250) — matrix operand order repaired and checked by reading, not compiled, not run on hardware
//
#include <hip/hip_runtime.h>
#include <math.h>

// Problem constants (match reference)
#define B_    128
#define T_    196
#define D_    512
#define TPAD  208          // 13 tiles of 16 rows, zero-padded
#define NRT   13           // row/col tiles per batch
#define NTILE 26           // 13 col tiles x {Q·Qt, Q·K2t}
#define INV_TEMP 5.0f      // 1 / 0.2
#define WPB_SIM 4          // waves per block in the WMMA kernel (64KB LDS)

typedef __attribute__((ext_vector_type(16))) _Float16 v16h;
typedef __attribute__((ext_vector_type(8)))  _Float16 v8h;
typedef __attribute__((ext_vector_type(8)))  float    v8f;
typedef __attribute__((ext_vector_type(4)))  float    v4f;

union Frag { v16h v; v8h ch[2]; };

__device__ __forceinline__ float wave_sum(float v) {
#pragma unroll
  for (int o = 16; o > 0; o >>= 1) v += __shfl_xor(v, o, 32);
  return v;
}
// exp(a-b) safe when a==b==-inf (returns 1, multiplied by s==0 -> 0)
__device__ __forceinline__ float sexp(float a, float b) {
  return (a == b) ? 1.0f : __expf(a - b);
}

// ---------------------------------------------------------------------------
// Kernel 1: row sums of both label matrices (idx1/idx2 predicates).
// ---------------------------------------------------------------------------
__global__ void k_rowsums(const float* __restrict__ L1, const float* __restrict__ L2,
                          float* __restrict__ rs1, float* __restrict__ rs2) {
  int g = blockIdx.x * (blockDim.x >> 5) + (threadIdx.x >> 5);
  int lane = threadIdx.x & 31;
  if (g >= B_ * T_) return;
  const float* r1 = L1 + (size_t)g * T_;
  const float* r2 = L2 + (size_t)g * T_;
  float a = 0.0f, b = 0.0f;
  for (int i = lane; i < T_; i += 32) { a += r1[i]; b += r2[i]; }
  a = wave_sum(a); b = wave_sum(b);
  if (lane == 0) { rs1[g] = a; rs2[g] = b; }
}

// ---------------------------------------------------------------------------
// Kernel 2: zero the per-batch CE accumulators (ws is poisoned by harness).
// ---------------------------------------------------------------------------
__global__ void k_zero(float* __restrict__ ce) {
  if ((int)threadIdx.x < B_) ce[threadIdx.x] = 0.0f;
}

// ---------------------------------------------------------------------------
// Kernel 3: L2-normalize crop1/crop2 into padded f16 arrays [B][TPAD][D].
// ---------------------------------------------------------------------------
__global__ void k_norm(const float* __restrict__ c1, const float* __restrict__ c2,
                       _Float16* __restrict__ q16, _Float16* __restrict__ k16) {
  int g = blockIdx.x * (blockDim.x >> 5) + (threadIdx.x >> 5);
  int lane = threadIdx.x & 31;
  const int nslot = B_ * TPAD;
  if (g >= 2 * nslot) return;
  const float* src = (g < nslot) ? c1 : c2;
  _Float16* dst    = (g < nslot) ? q16 : k16;
  int s = (g < nslot) ? g : g - nslot;
  int b = s / TPAD, t = s % TPAD;
  _Float16* drow = dst + ((size_t)b * TPAD + t) * D_ + lane * 16;
  if (t >= T_) {
    v8h z = { (_Float16)0, (_Float16)0, (_Float16)0, (_Float16)0,
              (_Float16)0, (_Float16)0, (_Float16)0, (_Float16)0 };
    *(v8h*)drow = z; *(v8h*)(drow + 8) = z;
    return;
  }
  const float* srow = src + ((size_t)b * T_ + t) * D_ + lane * 16;
  float x[16]; float ss = 0.0f;
#pragma unroll
  for (int i = 0; i < 16; i += 4) {
    v4f v = *(const v4f*)(srow + i);
    x[i] = v.x; x[i + 1] = v.y; x[i + 2] = v.z; x[i + 3] = v.w;
    ss += v.x * v.x + v.y * v.y + v.z * v.z + v.w * v.w;
  }
  ss = wave_sum(ss);
  float inv = 1.0f / fmaxf(sqrtf(ss), 1e-12f);   // matches F.normalize eps
  v8h o0, o1;
#pragma unroll
  for (int i = 0; i < 8; i++) {
    o0[i] = (_Float16)(x[i] * inv);
    o1[i] = (_Float16)(x[8 + i] * inv);
  }
  *(v8h*)drow = o0; *(v8h*)(drow + 8) = o1;
}

// ---------------------------------------------------------------------------
// Kernel 4: sparse masked-average positive key + positive logit (fp32).
// The reference's /colsum scaling cancels under L2-normalize, so it drops.
// ---------------------------------------------------------------------------
__global__ void k_pos(const float* __restrict__ c1, const float* __restrict__ c2,
                      const float* __restrict__ L1, float* __restrict__ pos) {
  int g = blockIdx.x * (blockDim.x >> 5) + (threadIdx.x >> 5);
  int lane = threadIdx.x & 31;
  if (g >= B_ * T_) return;
  int b = g / T_, t = g % T_;
  const float* lrow = L1 + ((size_t)b * T_ + t) * T_;
  float acc[16];
#pragma unroll
  for (int i = 0; i < 16; i++) acc[i] = 0.0f;
  for (int s0 = 0; s0 < T_; s0 += 32) {
    int s = s0 + lane;
    float lab = (s < T_) ? lrow[s] : 0.0f;
    unsigned long long m = __ballot(lab != 0.0f);
    while (m) {
      int j = __ffsll(m) - 1;
      m &= m - 1;
      const float* krow = c2 + ((size_t)b * T_ + (s0 + j)) * D_ + lane * 16;
#pragma unroll
      for (int i = 0; i < 16; i += 4) {
        v4f v = *(const v4f*)(krow + i);
        acc[i] += v.x; acc[i + 1] += v.y; acc[i + 2] += v.z; acc[i + 3] += v.w;
      }
    }
  }
  const float* qrow = c1 + ((size_t)b * T_ + t) * D_ + lane * 16;
  float ssq = 0.0f, ssp = 0.0f, dot = 0.0f;
#pragma unroll
  for (int i = 0; i < 16; i += 4) {
    v4f q = *(const v4f*)(qrow + i);
    ssq += q.x * q.x + q.y * q.y + q.z * q.z + q.w * q.w;
    ssp += acc[i] * acc[i] + acc[i + 1] * acc[i + 1] +
           acc[i + 2] * acc[i + 2] + acc[i + 3] * acc[i + 3];
    dot += q.x * acc[i] + q.y * acc[i + 1] + q.z * acc[i + 2] + q.w * acc[i + 3];
  }
  ssq = wave_sum(ssq); ssp = wave_sum(ssp); dot = wave_sum(dot);
  if (lane == 0)
    pos[g] = dot / (fmaxf(sqrtf(ssq), 1e-12f) * fmaxf(sqrtf(ssp), 1e-12f));
}

// ---------------------------------------------------------------------------
// WMMA helper: process NT consecutive output tiles.
//  * A fragments reload from LDS each k-step; the opaque asm is applied to a
//    32-bit OFFSET (not the pointer) so addrspace inference still yields
//    ds_load_b128 (DScnt, decoupled from LOADcnt) while LICM cannot hoist
//    the loads into 128 live VGPRs.
//  * sched_group_barrier pins a prefetch-distance-1 pipeline:
//    (VMEM 2NT) [(VMEM 2NT)(DS 2)(WMMA NT)]x15 (DS 2)(WMMA NT)
// ---------------------------------------------------------------------------
template <int NT>
__device__ __forceinline__ void do_group(
    int jbase, int b, int n, int h, int lane,
    const _Float16* __restrict__ Qb, const _Float16* __restrict__ Kb,
    const float* __restrict__ rs1, const float* __restrict__ rs2,
    const _Float16* Alds, float* m, float* s) {
  unsigned aoff = 0;
  asm volatile("" : "+v"(aoff));              // opaque 0: blocks LICM hoisting
  const _Float16* Ap = Alds + aoff;           // base still traceable to LDS
  const _Float16* brow[NT];
  bool neg[NT];
  v8f c[NT];
  const v8f vzero = { 0.f, 0.f, 0.f, 0.f, 0.f, 0.f, 0.f, 0.f };
#pragma unroll
  for (int t = 0; t < NT; t++) {
    int j = jbase + t;                        // tile j: pass=j/13, ct=j%13
    int p = j / NRT, ct = j % NRT;
    const _Float16* Y = p ? Kb : Qb;
    const float* rsp = p ? rs2 : rs1;
    int col = ct * 16 + n;
    neg[t] = (col < T_) && (rsp[b * T_ + col] < 1.0f);
    brow[t] = Y + (size_t)col * D_ + 8 * h;
    c[t] = vzero;
  }
#pragma unroll
  for (int i = 0; i < 16; i++) {              // kb = 32*i
    Frag bb[NT];
#pragma unroll
    for (int t = 0; t < NT; t++) {            // 2*NT loads
      bb[t].ch[0] = *(const v8h*)(brow[t] + i * 32);
      bb[t].ch[1] = *(const v8h*)(brow[t] + i * 32 + 16);
    }
    Frag ua;
    ua.ch[0] = *(const v8h*)(Ap + i * 512 + lane * 8);
    ua.ch[1] = *(const v8h*)(Ap + i * 512 + 256 + lane * 8);
#pragma unroll
    for (int t = 0; t < NT; t++)
      c[t] = __builtin_amdgcn_wmma_f32_16x16x32_f16(false, ua.v, false, bb[t].v,
                                                    (short)0, c[t], false, false);
  }
#if __has_builtin(__builtin_amdgcn_sched_group_barrier)
  // software pipeline, prefetch distance 1
  __builtin_amdgcn_sched_group_barrier(0x020, 2 * NT, 0);   // B loads, step 0
#pragma unroll
  for (int i = 0; i < 16; i++) {
    if (i < 15)
      __builtin_amdgcn_sched_group_barrier(0x020, 2 * NT, 0); // B loads, i+1
    __builtin_amdgcn_sched_group_barrier(0x100, 2, 0);        // A ds reads, i
    __builtin_amdgcn_sched_group_barrier(0x008, NT, 0);       // WMMAs, step i
  }
#endif
#pragma unroll
  for (int t = 0; t < NT; t++) {
    if (neg[t]) {
#pragma unroll
      for (int v = 0; v < 8; v++) {
        float lg = c[t][v] * INV_TEMP;
        float nm = fmaxf(m[v], lg);
        s[v] = s[v] * sexp(m[v], nm) + sexp(lg, nm);
        m[v] = nm;
      }
    }
  }
}

// ---------------------------------------------------------------------------
// Kernel 5 (WMMA): fused similarity matmuls + masked online logsumexp + CE.
// One wave per (batch, 16-row tile); A row-tile staged via LDS in fragment
// layout; 26 output tiles processed as 6 groups of 4 + 1 group of 2.
// C-layout: lane l -> col = l&15, rows = v + 8*(l>>4), v=0..7.
// A f16 layout: lane row = l&15; chunks at K = kb+8h and kb+16+8h.
// B f16 layout: lane col = l&15; chunks at K = kb+8h and kb+16+8h.
// ---------------------------------------------------------------------------
__global__ __launch_bounds__(WPB_SIM * 32)
void k_sim_lse(const _Float16* __restrict__ q16, const _Float16* __restrict__ k16,
               const float* __restrict__ rs1, const float* __restrict__ rs2,
               const float* __restrict__ pos, float* __restrict__ ce_sum) {
  // per wave: 16 kblocks * 2 chunks * 32 lanes * 8 halves = 8192 halves = 16KB
  __shared__ _Float16 ldsA[WPB_SIM * 8192];
  int w = threadIdx.x >> 5;
  int lane = threadIdx.x & 31;
  int g = blockIdx.x * WPB_SIM + w;
  if (g >= B_ * NRT) return;
  int b = g / NRT, rt = g % NRT;
  int rowbase = rt * 16;
  int n = lane & 15, h = lane >> 4;
  const _Float16* Qb = q16 + (size_t)b * TPAD * D_;
  const _Float16* Kb = k16 + (size_t)b * TPAD * D_;
  _Float16* Alds = ldsA + w * 8192;

  // ---- stage this wave's A row-tile into LDS in fragment layout ----
  {
    const _Float16* arow = Qb + (size_t)(rowbase + n) * D_;
#pragma unroll
    for (int i = 0; i < 16; i++) {
      v8h c0 = *(const v8h*)(arow + i * 32 + 8 * h);        // K = kb+8h .. +8
      v8h c1 = *(const v8h*)(arow + i * 32 + 16 + 8 * h);   // K = kb+16+8h ..
      *(v8h*)(Alds + i * 512 + lane * 8)       = c0;
      *(v8h*)(Alds + i * 512 + 256 + lane * 8) = c1;
    }
  }
  // same-wave LDS ops are in order; no cross-wave sharing -> no barrier

  float m[8], s[8];
#pragma unroll
  for (int v = 0; v < 8; v++) { m[v] = -__builtin_inff(); s[v] = 0.0f; }

  // ---- 26 output tiles: 6 groups of 4, then a tail group of 2 ----
#pragma unroll 1
  for (int gi = 0; gi < 6; gi++)
    do_group<4>(gi * 4, b, n, h, lane, Qb, Kb, rs1, rs2, Alds, m, s);
  do_group<2>(24, b, n, h, lane, Qb, Kb, rs1, rs2, Alds, m, s);

  // merge (m,s) across the 16 columns held by each half-wave
#pragma unroll
  for (int o = 1; o < 16; o <<= 1) {
#pragma unroll
    for (int v = 0; v < 8; v++) {
      float om = __shfl_xor(m[v], o, 32);
      float os = __shfl_xor(s[v], o, 32);
      float nm = fmaxf(m[v], om);
      s[v] = s[v] * sexp(m[v], nm) + os * sexp(om, nm);
      m[v] = nm;
    }
  }
  if (n == 0) {   // lanes 0 and 16 hold final per-row states (rows +8h)
    float ce = 0.0f;
#pragma unroll
    for (int v = 0; v < 8; v++) {
      int row = rowbase + 8 * h + v;
      if (row < T_ && rs1[b * T_ + row] >= 1.0f) {
        float l0 = pos[b * T_ + row] * INV_TEMP;      // positive logit
        float nm = fmaxf(m[v], l0);
        float st = s[v] * sexp(m[v], nm) + sexp(l0, nm);
        ce += nm + __logf(st) - l0;                    // LSE - logits[0]
      }
    }
    atomicAdd(&ce_sum[b], ce);
  }
}

// ---------------------------------------------------------------------------
// Kernel 6: finalize — per-batch mean over positives, global have_negs gate.
// ---------------------------------------------------------------------------
__global__ void k_final(const float* __restrict__ rs1, const float* __restrict__ rs2,
                        const float* __restrict__ ce_sum, float* __restrict__ out) {
  __shared__ float sh_loss[B_];
  __shared__ int sh_n1[B_], sh_n2[B_];
  int b = threadIdx.x;
  if (b < B_) {
    int cnt = 0, nn1 = 0, nn2 = 0;
    for (int t = 0; t < T_; t++) {
      if (rs1[b * T_ + t] >= 1.0f) cnt++; else nn1++;
      if (rs2[b * T_ + t] < 1.0f) nn2++;
    }
    sh_loss[b] = (cnt > 0) ? ce_sum[b] / (float)cnt : 0.0f;
    sh_n1[b] = nn1; sh_n2[b] = nn2;
  }
  __syncthreads();
  if (b == 0) {
    float tot = 0.0f; int a1 = 0, a2 = 0;
    for (int i = 0; i < B_; i++) { tot += sh_loss[i]; a1 += sh_n1[i]; a2 += sh_n2[i]; }
    out[0] = (a1 > 0 && a2 > 0) ? tot / (float)B_ : 0.0f;
  }
}

// ---------------------------------------------------------------------------
extern "C" void kernel_launch(void* const* d_in, const int* in_sizes, int n_in,
                              void* d_out, int out_size, void* d_ws, size_t ws_size,
                              hipStream_t stream) {
  const float* crop1 = (const float*)d_in[0];
  const float* crop2 = (const float*)d_in[1];
  const float* L1    = (const float*)d_in[2];   // bce_labelsl2s
  const float* L2    = (const float*)d_in[3];   // bce_labelss2l
  float* out = (float*)d_out;

  // Workspace layout (~54.9 MB total)
  char* ws = (char*)d_ws;
  size_t off = 0;
  _Float16* q16 = (_Float16*)(ws + off); off += (size_t)B_ * TPAD * D_ * 2;  // 27.3 MB
  _Float16* k16 = (_Float16*)(ws + off); off += (size_t)B_ * TPAD * D_ * 2;  // 27.3 MB
  float* rs1    = (float*)(ws + off);    off += (size_t)B_ * T_ * 4;
  float* rs2    = (float*)(ws + off);    off += (size_t)B_ * T_ * 4;
  float* pos    = (float*)(ws + off);    off += (size_t)B_ * T_ * 4;
  float* ce_sum = (float*)(ws + off);    off += (size_t)B_ * 4;
  (void)ws_size; (void)in_sizes; (void)n_in; (void)out_size;

  const int WPB = 8;  // waves per 256-thread block for the simple kernels

  k_zero<<<1, 128, 0, stream>>>(ce_sum);

  int rows = B_ * T_;                                     // 25088
  k_rowsums<<<(rows + WPB - 1) / WPB, 256, 0, stream>>>(L1, L2, rs1, rs2);

  int slots = 2 * B_ * TPAD;                              // 53248
  k_norm<<<(slots + WPB - 1) / WPB, 256, 0, stream>>>(crop1, crop2, q16, k16);

  k_pos<<<(rows + WPB - 1) / WPB, 256, 0, stream>>>(crop1, crop2, L1, pos);

  int tiles = B_ * NRT;                                   // 1664 waves
  k_sim_lse<<<(tiles + WPB_SIM - 1) / WPB_SIM, WPB_SIM * 32, 0, stream>>>(
      q16, k16, rs1, rs2, pos, ce_sum);

  k_final<<<1, 128, 0, stream>>>(rs1, rs2, ce_sum, out);
}
